// BRFCell_2499670966836
// MI455X (gfx1250) — compile-verified, matrix operand order checked by reading, not run on hardware
//
#include <hip/hip_runtime.h>

typedef __attribute__((ext_vector_type(16))) __bf16 v16bf;
typedef __attribute__((ext_vector_type(8)))  __bf16 v8bf;
typedef __attribute__((ext_vector_type(8)))  float  v8f;

#define BB 4096   // batch
#define KK 256    // input dim
#define NN 4096   // neurons
#define DT 0.01f

// round-to-nearest-even f32 -> bf16 (bit manipulation; no reliance on cast lowering)
__device__ __forceinline__ __bf16 f2bf(float f) {
  unsigned u = __float_as_uint(f);
  u += 0x7FFFu + ((u >> 16) & 1u);
  unsigned short h = (unsigned short)(u >> 16);
  return __builtin_bit_cast(__bf16, h);
}

// Convert x and W (each BB*KK == NN*KK == 1M f32) to bf16 in workspace.
__global__ void __launch_bounds__(256) cvt_kernel(const float* __restrict__ x,
                                                  const float* __restrict__ w,
                                                  __bf16* __restrict__ xb,
                                                  __bf16* __restrict__ wb) {
  int i = blockIdx.x * 256 + threadIdx.x;   // grid sized exactly: no bounds check
  xb[i] = f2bf(x[i]);
  wb[i] = f2bf(w[i]);
}

// Load one wave32 bf16 16x32 A-fragment (or 32x16 B-fragment: identical pattern,
// since B column n == W row n, K contiguous in memory for both).
// Lane r (r = lane&15), half = lane>>4:
//   VGPRs 0-3: K = k0 + half*8 .. +7      (one b128 load)
//   VGPRs 4-7: K = k0 + 16 + half*8 .. +7 (one b128 load)
__device__ __forceinline__ v16bf load_frag(const __bf16* __restrict__ rowp, int k0, int half) {
  v8bf lo = *reinterpret_cast<const v8bf*>(rowp + k0 + (half << 3));
  v8bf hi = *reinterpret_cast<const v8bf*>(rowp + k0 + 16 + (half << 3));
  v16bf r;
#pragma unroll
  for (int i = 0; i < 8; ++i) { r[i] = lo[i]; r[i + 8] = hi[i]; }
  return r;
}

__global__ void __launch_bounds__(256) brf_kernel(
    const __bf16* __restrict__ xb, const __bf16* __restrict__ wb,
    const float* __restrict__ z, const float* __restrict__ u,
    const float* __restrict__ v, const float* __restrict__ q,
    const float* __restrict__ omega_p, const float* __restrict__ boff_p,
    float* __restrict__ out) {
  const int lane = threadIdx.x & 31;
  const int warp = threadIdx.x >> 5;
  const int wid  = blockIdx.x * 8 + warp;      // 16384 waves total
  const int m0   = (wid >> 6) << 4;            // 256 M-tiles of 16 rows
  const int n0   = (wid & 63) << 6;            // 64 N-supertiles of 64 cols
  const int r    = lane & 15;
  const int half = lane >> 4;

  v8f acc[4];
#pragma unroll
  for (int t = 0; t < 4; ++t)
#pragma unroll
    for (int i = 0; i < 8; ++i) acc[t][i] = 0.0f;

  const __bf16* arow = xb + (size_t)(m0 + r) * KK;
#pragma unroll
  for (int kc = 0; kc < KK; kc += 32) {
    v16bf af = load_frag(arow, kc, half);
#pragma unroll
    for (int t = 0; t < 4; ++t) {
      const __bf16* brow = wb + (size_t)(n0 + t * 16 + r) * KK;
      v16bf bfr = load_frag(brow, kc, half);
      acc[t] = __builtin_amdgcn_wmma_f32_16x16x32_bf16(
          false, af, false, bfr, (short)0, acc[t], false, false);
    }
  }

  const size_t BN = (size_t)BB * NN;
  float* out_z = out;
  float* out_u = out + BN;
  float* out_v = out + 2 * BN;
  float* out_q = out + 3 * BN;

  // C/D layout: lane n-index = lane&15; VGPR j holds row m0 + 8*half + j.
#pragma unroll
  for (int t = 0; t < 4; ++t) {
    const int n = n0 + t * 16 + r;
    const float om   = fabsf(omega_p[n]);
    const float dom  = DT * om;
    const float p_om = (-1.0f + sqrtf(1.0f - dom * dom)) * (1.0f / DT);
    const float bo   = fabsf(boff_p[n]);
    const float cw   = __cosf(dom);
    const float sw   = __sinf(dom);
#pragma unroll
    for (int j = 0; j < 8; ++j) {
      const int m = m0 + (half << 3) + j;
      const size_t idx = (size_t)m * NN + n;
      const float uo = __builtin_nontemporal_load(u + idx);
      const float vo = __builtin_nontemporal_load(v + idx);
      const float qo = __builtin_nontemporal_load(q + idx);
      const float zo = __builtin_nontemporal_load(z + idx);
      // b = p_omega - |b_offset| - q - q  (q subtracted twice, faithful to ref)
      const float bb = p_om - bo - 2.0f * qo;
      const float e  = __expf(bb * DT);
      const float un = e * (uo * cw - vo * sw) + acc[t][j] * DT;
      const float vn = e * (uo * sw + vo * cw);
      const float qn = 0.9f * qo + zo;
      const float zn = (un - 1.0f - qn) > 0.0f ? 1.0f : 0.0f;
      __builtin_nontemporal_store(zn, out_z + idx);
      __builtin_nontemporal_store(un, out_u + idx);
      __builtin_nontemporal_store(vn, out_v + idx);
      __builtin_nontemporal_store(qn, out_q + idx);
    }
  }
}

extern "C" void kernel_launch(void* const* d_in, const int* in_sizes, int n_in,
                              void* d_out, int out_size, void* d_ws, size_t ws_size,
                              hipStream_t stream) {
  const float* x  = (const float*)d_in[0];
  const float* z  = (const float*)d_in[1];
  const float* u  = (const float*)d_in[2];
  const float* v  = (const float*)d_in[3];
  const float* q  = (const float*)d_in[4];
  const float* W  = (const float*)d_in[5];
  const float* om = (const float*)d_in[6];
  const float* bo = (const float*)d_in[7];
  float* out = (float*)d_out;

  __bf16* xb = (__bf16*)d_ws;                 // 2 MB
  __bf16* wb = xb + (size_t)BB * KK;          // 2 MB

  cvt_kernel<<<(BB * KK) / 256, 256, 0, stream>>>(x, W, xb, wb);

  const int waves  = (BB / 16) * (NN / 64);   // 16384
  const int blocks = waves / 8;               // 2048 blocks of 8 wave32s
  brf_kernel<<<blocks, 256, 0, stream>>>(xb, wb, z, u, v, q, om, bo, out);
}
